// SurMoE_7215545057866
// MI455X (gfx1250) — compile-verified, matrix operand order checked by reading, not run on hardware
//
#include <hip/hip_runtime.h>
#include <math.h>
#include <stdint.h>

typedef __attribute__((ext_vector_type(16))) _Float16 v16h;
typedef __attribute__((ext_vector_type(8)))  _Float16 h8;
typedef __attribute__((ext_vector_type(8)))  float    v8f;
typedef __attribute__((ext_vector_type(4)))  float    f4;

constexpr int Bv = 32, N1v = 512, N2v = 256, Dv = 256, Hv = 8, Sv = 768;

// ---------------------------------------------------------------------------
// CDNA5 async global->LDS copies (ASYNCcnt), per-lane addresses.
// LDS offset = low 32 bits of the generic shared pointer (LDS aperture keeps
// the offset in addr[31:0] per ISA §10.2).
// ---------------------------------------------------------------------------
__device__ __forceinline__ uint32_t lds_off(const void* p) {
    return (uint32_t)(uintptr_t)p;
}
__device__ __forceinline__ void async_b128(uint32_t lds, const void* g) {
    asm volatile("global_load_async_to_lds_b128 %0, %1, off"
                 :: "v"(lds), "v"((unsigned long long)(uintptr_t)g) : "memory");
}
__device__ __forceinline__ void async_b32(uint32_t lds, const void* g) {
    asm volatile("global_load_async_to_lds_b32 %0, %1, off"
                 :: "v"(lds), "v"((unsigned long long)(uintptr_t)g) : "memory");
}
__device__ __forceinline__ void wait_async0() {
    asm volatile("s_wait_asynccnt 0x0" ::: "memory");
}

// Build a v16h WMMA operand from an f32 LDS row (A-layout k mapping):
// elem j<8 -> row[kb8+j], j>=8 -> row[16+kb8+(j-8)].  4 aligned ds_load_b128.
__device__ __forceinline__ v16h frag_f32(const float* row, int kb8, float scale) {
    const f4* r4 = (const f4*)row;
    int q = kb8 >> 2;                 // 0 or 2
    f4 x0 = r4[q], x1 = r4[q + 1], x2 = r4[q + 4], x3 = r4[q + 5];
    v16h f;
    f[0]=(_Float16)(x0.x*scale); f[1]=(_Float16)(x0.y*scale); f[2]=(_Float16)(x0.z*scale); f[3]=(_Float16)(x0.w*scale);
    f[4]=(_Float16)(x1.x*scale); f[5]=(_Float16)(x1.y*scale); f[6]=(_Float16)(x1.z*scale); f[7]=(_Float16)(x1.w*scale);
    f[8]=(_Float16)(x2.x*scale); f[9]=(_Float16)(x2.y*scale); f[10]=(_Float16)(x2.z*scale); f[11]=(_Float16)(x2.w*scale);
    f[12]=(_Float16)(x3.x*scale); f[13]=(_Float16)(x3.y*scale); f[14]=(_Float16)(x3.z*scale); f[15]=(_Float16)(x3.w*scale);
    return f;
}

// ---------------------------------------------------------------------------
// Router (tiny: one block per batch, plain VALU)
// ---------------------------------------------------------------------------
__global__ __launch_bounds__(256) void router_kernel(
    const float* __restrict__ x1, const float* __restrict__ x2,
    const float* __restrict__ r_w1, const float* __restrict__ r_b1,
    const float* __restrict__ ln_g, const float* __restrict__ ln_b,
    const float* __restrict__ r_w2, const float* __restrict__ r_b2,
    float* __restrict__ wout)
{
    int b = blockIdx.x, t = threadIdx.x;
    __shared__ float m[2 * Dv];
    __shared__ float red[256];
    __shared__ float logits[4];

    float s = 0.f;
    const float* p1 = x1 + (size_t)b * N1v * Dv + t;
    for (int n = 0; n < N1v; ++n) s += p1[(size_t)n * Dv];
    m[t] = s * (1.f / N1v);
    s = 0.f;
    const float* p2 = x2 + (size_t)b * N2v * Dv + t;
    for (int n = 0; n < N2v; ++n) s += p2[(size_t)n * Dv];
    m[Dv + t] = s * (1.f / N2v);
    __syncthreads();

    float y = r_b1[t];
    const float* wr = r_w1 + (size_t)t * (2 * Dv);
    for (int k = 0; k < 2 * Dv; ++k) y += m[k] * wr[k];

    red[t] = y; __syncthreads();
    for (int o = 128; o; o >>= 1) { if (t < o) red[t] += red[t + o]; __syncthreads(); }
    float mu = red[0] * (1.f / Dv);
    __syncthreads();
    float dv = y - mu;
    red[t] = dv * dv; __syncthreads();
    for (int o = 128; o; o >>= 1) { if (t < o) red[t] += red[t + o]; __syncthreads(); }
    float var = red[0] * (1.f / Dv);
    __syncthreads();

    float hn = dv * rsqrtf(var + 1e-5f) * ln_g[t] + ln_b[t];
    float g  = 0.5f * hn * (1.f + erff(hn * 0.70710678118654752f));  // exact gelu

    for (int e = 0; e < 4; ++e) {
        red[t] = g * r_w2[(size_t)e * Dv + t]; __syncthreads();
        for (int o = 128; o; o >>= 1) { if (t < o) red[t] += red[t + o]; __syncthreads(); }
        if (t == 0) logits[e] = red[0] + r_b2[e];
        __syncthreads();
    }
    if (t == 0) {
        float mx = fmaxf(fmaxf(logits[0], logits[1]), fmaxf(logits[2], logits[3]));
        float e0 = expf(logits[0] - mx), e1 = expf(logits[1] - mx);
        float e2 = expf(logits[2] - mx), e3 = expf(logits[3] - mx);
        float inv = 1.f / (e0 + e1 + e2 + e3);
        wout[b * 4 + 0] = e0 * inv; wout[b * 4 + 1] = e1 * inv;
        wout[b * 4 + 2] = e2 * inv; wout[b * 4 + 3] = e3 * inv;
    }
}

// ---------------------------------------------------------------------------
// Generic WMMA GEMM with async double-buffered LDS tiles.
//   C = act(A @ B' + bias) (optionally +=)
//   A: (M,K) row-major f32.  BT=true: B is (N,K); BT=false: B is (K,N)
//   (transposed into LDS [n][k] by per-lane async b32 copies).
//   Block tile 128(M) x 64(N), 8 waves in 4x2; each wave computes 32x32 via
//   four v_wmma_f32_16x16x32_f16 (2 A-frags x 2 B-frags reused).
// ---------------------------------------------------------------------------
template<int ACT, bool BT, bool BIASM, bool ACC>
__global__ __launch_bounds__(256) void gemm_wmma_kernel(
    const float* __restrict__ A, int lda, long long sAb,
    const float* __restrict__ Bw, int ldb, long long sBb,
    const float* __restrict__ bias,
    float* __restrict__ C, int ldc, long long sCb,
    int M, int N, int K)
{
    __shared__ __align__(16) float As[2][128][36];  // [m][k]
    __shared__ __align__(16) float Bs[2][64][36];   // [n][k]

    int t = threadIdx.x, lane = t & 31, wave = t >> 5;
    int bm = blockIdx.y, bn = blockIdx.x, bz = blockIdx.z;
    const float* Ab = A  + (long long)bz * sAb;
    const float* Bb = Bw + (long long)bz * sBb;
    float*       Cb = C  + (long long)bz * sCb;

    int waveM = wave >> 1;          // 0..3 -> 32-row sub-tile
    int waveN = wave & 1;           // 0..1 -> 32-col sub-tile
    v8f c00 = {}, c01 = {}, c10 = {}, c11 = {};

    int kb8   = (lane & 16) ? 8 : 0;
    int lrowA = t >> 1;             // 0..127
    int lkcA  = (t & 1) * 16;       // 0 or 16
    int lrowB = t >> 2;             // 0..63
    int lkcB  = (t & 3) * 8;        // 0,8,16,24

    auto issue = [&](int buf, int kk) {
        const float* ag = Ab + (size_t)(bm * 128 + lrowA) * lda + kk + lkcA;
        async_b128(lds_off(&As[buf][lrowA][lkcA]),      ag);
        async_b128(lds_off(&As[buf][lrowA][lkcA + 4]),  ag + 4);
        async_b128(lds_off(&As[buf][lrowA][lkcA + 8]),  ag + 8);
        async_b128(lds_off(&As[buf][lrowA][lkcA + 12]), ag + 12);
        if (BT) {
            const float* bg = Bb + (size_t)(bn * 64 + lrowB) * ldb + kk + lkcB;
            async_b128(lds_off(&Bs[buf][lrowB][lkcB]),     bg);
            async_b128(lds_off(&Bs[buf][lrowB][lkcB + 4]), bg + 4);
        } else {
            #pragma unroll
            for (int j = 0; j < 8; ++j)
                async_b32(lds_off(&Bs[buf][lrowB][lkcB + j]),
                          Bb + (size_t)(kk + lkcB + j) * ldb + bn * 64 + lrowB);
        }
    };

    issue(0, 0);
    int nk = K >> 5;
    for (int i = 0; i < nk; ++i) {
        wait_async0();
        __syncthreads();
        if (i + 1 < nk) issue((i + 1) & 1, (i + 1) << 5);

        const float* arow0 = &As[i & 1][waveM * 32 + (lane & 15)][0];
        const float* arow1 = arow0 + 16 * 36;
        const float* brow0 = &Bs[i & 1][waveN * 32 + (lane & 15)][0];
        const float* brow1 = brow0 + 16 * 36;
        v16h af0 = frag_f32(arow0, kb8, 1.f);
        v16h af1 = frag_f32(arow1, kb8, 1.f);
        v16h bf0 = frag_f32(brow0, kb8, 1.f);
        v16h bf1 = frag_f32(brow1, kb8, 1.f);
        c00 = __builtin_amdgcn_wmma_f32_16x16x32_f16(false, af0, false, bf0, (short)0, c00, false, false);
        c01 = __builtin_amdgcn_wmma_f32_16x16x32_f16(false, af0, false, bf1, (short)0, c01, false, false);
        c10 = __builtin_amdgcn_wmma_f32_16x16x32_f16(false, af1, false, bf0, (short)0, c10, false, false);
        c11 = __builtin_amdgcn_wmma_f32_16x16x32_f16(false, af1, false, bf1, (short)0, c11, false, false);
        __syncthreads();
    }

    int nl0 = bn * 64 + waveN * 32 + (lane & 15);
    int half8 = (lane & 16) ? 8 : 0;
    #pragma unroll
    for (int rb = 0; rb < 2; ++rb) {
        v8f* cl = (rb == 0) ? &c00 : &c10;
        v8f* ch = (rb == 0) ? &c01 : &c11;
        #pragma unroll
        for (int v = 0; v < 8; ++v) {
            int mg = bm * 128 + waveM * 32 + rb * 16 + v + half8;
            float x0, x1v;
            if (BIASM) {
                float bb = bias[mg];
                x0 = (*cl)[v] + bb; x1v = (*ch)[v] + bb;
            } else {
                x0 = (*cl)[v] + bias[nl0]; x1v = (*ch)[v] + bias[nl0 + 16];
            }
            if (ACT == 1) {                          // ELU
                x0  = (x0  > 0.f) ? x0  : (expf(x0)  - 1.f);
                x1v = (x1v > 0.f) ? x1v : (expf(x1v) - 1.f);
            }
            float* crow = Cb + (size_t)mg * ldc;
            if (ACC) { crow[nl0] += x0; crow[nl0 + 16] += x1v; }
            else     { crow[nl0]  = x0; crow[nl0 + 16]  = x1v; }
        }
    }
}

// ---------------------------------------------------------------------------
// FlashAttention-2: qkv (B*S, 3*D); head h = cols h*32..+31 of each third.
// HD=32 == WMMA K.  8 waves/block, each wave owns a 16-row Q tile.
// ---------------------------------------------------------------------------
__global__ __launch_bounds__(256) void flash_kernel(
    const float* __restrict__ qkv, float* __restrict__ obuf)
{
    __shared__ __align__(16) _Float16 pbuf[8][16][40];   // per-wave P bounce

    int lane = threadIdx.x & 31, wave = threadIdx.x >> 5;
    int bh = blockIdx.y, b = bh >> 3, h = bh & 7;
    int q0 = (blockIdx.x * 8 + wave) * 16;
    size_t baseRow = (size_t)b * Sv;
    int kb8 = (lane & 16) ? 8 : 0;
    const float scale = 0.17677669529663687f;            // 1/sqrt(32)

    v16h qf = frag_f32(qkv + (baseRow + q0 + (lane & 15)) * (3 * Dv) + h * 32, kb8, scale);

    float mi[8], li[8];
    #pragma unroll
    for (int v = 0; v < 8; ++v) { mi[v] = -3.0e38f; li[v] = 0.f; }
    v8f o0 = {}, o1 = {};

    for (int kbb = 0; kbb < Sv; kbb += 32) {
        // scores = Q(16x32) @ K^T : two 16x16 tiles (contraction = hd)
        const float* kp0 = qkv + (baseRow + kbb + (lane & 15)) * (3 * Dv) + Dv + h * 32;
        v16h kf0 = frag_f32(kp0,                 kb8, 1.f);
        v16h kf1 = frag_f32(kp0 + 16 * (3 * Dv), kb8, 1.f);
        v8f z = {};
        v8f s0 = __builtin_amdgcn_wmma_f32_16x16x32_f16(false, qf, false, kf0, (short)0, z, false, false);
        v8f s1 = __builtin_amdgcn_wmma_f32_16x16x32_f16(false, qf, false, kf1, (short)0, z, false, false);

        // online softmax (rows live per (vgpr v, lane-half))
        float p0[8], p1[8];
        #pragma unroll
        for (int v = 0; v < 8; ++v) {
            float cm = fmaxf(s0[v], s1[v]);
            #pragma unroll
            for (int msk = 1; msk < 16; msk <<= 1) cm = fmaxf(cm, __shfl_xor(cm, msk, 32));
            float nm = fmaxf(mi[v], cm);
            float corr = expf(mi[v] - nm);
            float a0 = expf(s0[v] - nm), a1 = expf(s1[v] - nm);
            float rs = a0 + a1;
            #pragma unroll
            for (int msk = 1; msk < 16; msk <<= 1) rs += __shfl_xor(rs, msk, 32);
            li[v] = li[v] * corr + rs;
            mi[v] = nm;
            o0[v] *= corr; o1[v] *= corr;
            p0[v] = a0; p1[v] = a1;
        }

        // P: C-layout -> A-layout via wave-private LDS bounce
        int half8 = (lane & 16) ? 8 : 0;
        int cn = lane & 15;
        #pragma unroll
        for (int v = 0; v < 8; ++v) {
            pbuf[wave][v + half8][cn]      = (_Float16)p0[v];
            pbuf[wave][v + half8][cn + 16] = (_Float16)p1[v];
        }
        v16h pf;
        {
            const _Float16* pr = &pbuf[wave][lane & 15][0];
            h8 lo = *(const h8*)(pr + kb8);
            h8 hi = *(const h8*)(pr + 16 + kb8);
            #pragma unroll
            for (int j = 0; j < 8; ++j) { pf[j] = lo[j]; pf[8 + j] = hi[j]; }
        }

        // O += P(16x32) @ V(32x32) : contraction = keys
        v16h vf0, vf1;
        {
            int hd0 = lane & 15;
            #pragma unroll
            for (int j = 0; j < 16; ++j) {
                int kx = (j < 8) ? (kb8 + j) : (16 + kb8 + (j - 8));
                const float* vp = qkv + (baseRow + kbb + kx) * (3 * Dv) + 2 * Dv + h * 32;
                vf0[j] = (_Float16)vp[hd0];
                vf1[j] = (_Float16)vp[hd0 + 16];
            }
        }
        o0 = __builtin_amdgcn_wmma_f32_16x16x32_f16(false, pf, false, vf0, (short)0, o0, false, false);
        o1 = __builtin_amdgcn_wmma_f32_16x16x32_f16(false, pf, false, vf1, (short)0, o1, false, false);
    }

    int half8 = (lane & 16) ? 8 : 0;
    int n = lane & 15;
    #pragma unroll
    for (int v = 0; v < 8; ++v) {
        float inv = 1.f / li[v];
        float* op = obuf + (baseRow + q0 + v + half8) * Dv + h * 32;
        op[n]      = o0[v] * inv;
        op[n + 16] = o1[v] * inv;
    }
}

// ---------------------------------------------------------------------------
// Final weighted combines
// ---------------------------------------------------------------------------
__global__ __launch_bounds__(256) void combine_gene_kernel(
    const float* __restrict__ lf, const float* __restrict__ xout,
    const float* __restrict__ ofin, const float* __restrict__ x1,
    const float* __restrict__ w, float* __restrict__ out)
{
    size_t idx = (size_t)blockIdx.x * blockDim.x + threadIdx.x;  // B*N1*D
    int d = idx % Dv;
    size_t nd = idx / Dv;
    int n = nd % N1v;
    int b = nd / N1v;
    size_t zr = ((size_t)b * Sv + n) * Dv + d;
    out[idx] = w[b*4+0]*lf[zr] + w[b*4+1]*xout[idx] + w[b*4+2]*ofin[zr] + w[b*4+3]*x1[idx];
}

__global__ __launch_bounds__(256) void combine_img_kernel(
    const float* __restrict__ lf, const float* __restrict__ xout,
    const float* __restrict__ ofin, const float* __restrict__ x2,
    const float* __restrict__ w, float* __restrict__ out)
{
    size_t idx = (size_t)blockIdx.x * blockDim.x + threadIdx.x;  // B*N2*D
    int d = idx % Dv;
    size_t nd = idx / Dv;
    int n = nd % N2v;
    int b = nd / N2v;
    size_t zr = ((size_t)b * Sv + N1v + n) * Dv + d;
    size_t x1r = ((size_t)b * N1v + n) * Dv + d;   // i1 = xout[:, :N2, :]
    out[idx] = w[b*4+0]*lf[zr] + w[b*4+1]*xout[x1r] + w[b*4+2]*ofin[zr] + w[b*4+3]*x2[idx];
}

// ---------------------------------------------------------------------------
extern "C" void kernel_launch(void* const* d_in, const int* in_sizes, int n_in,
                              void* d_out, int out_size, void* d_ws, size_t ws_size,
                              hipStream_t stream)
{
    const float* x1        = (const float*)d_in[0];
    const float* x2        = (const float*)d_in[1];
    const float* r_w1      = (const float*)d_in[2];
    const float* r_b1      = (const float*)d_in[3];
    const float* ln_g      = (const float*)d_in[4];
    const float* ln_b      = (const float*)d_in[5];
    const float* r_w2      = (const float*)d_in[6];
    const float* r_b2      = (const float*)d_in[7];
    const float* lf_w      = (const float*)d_in[8];
    const float* lf_b      = (const float*)d_in[9];
    const float* af2_w     = (const float*)d_in[10];
    const float* af2_b     = (const float*)d_in[11];
    const float* af3_w     = (const float*)d_in[12];
    const float* af3_b     = (const float*)d_in[13];
    const float* attn_in_w = (const float*)d_in[14];
    const float* attn_in_b = (const float*)d_in[15];
    const float* attn_out_w= (const float*)d_in[16];
    const float* attn_out_b= (const float*)d_in[17];

    float* gene = (float*)d_out;                       // (B,N1,D)
    float* img  = gene + (size_t)Bv * N1v * Dv;        // (B,N2,D)

    float* ws   = (float*)d_ws;
    float* lf   = ws;                                  size_t o = (size_t)Bv*Sv*Dv;
    float* xout = ws + o;                              o += (size_t)Bv*N1v*Dv;
    float* qkvb = ws + o;                              o += (size_t)Bv*Sv*3*Dv;
    float* obuf = ws + o;                              o += (size_t)Bv*Sv*Dv;
    float* ofin = ws + o;                              o += (size_t)Bv*Sv*Dv;
    float* wrt  = ws + o;                              o += (size_t)Bv*4;

    // 1) router weights
    router_kernel<<<Bv, 256, 0, stream>>>(x1, x2, r_w1, r_b1, ln_g, ln_b, r_w2, r_b2, wrt);

    // 2) lf = elu(z @ lf_w.T + b): two batched launches (x1 rows, x2 rows)
    gemm_wmma_kernel<1,true,false,false><<<dim3(Dv/64, N1v/128, Bv), 256, 0, stream>>>(
        x1, Dv, (long long)N1v*Dv, lf_w, Dv, 0, lf_b,
        lf, Dv, (long long)Sv*Dv, N1v, Dv, Dv);
    gemm_wmma_kernel<1,true,false,false><<<dim3(Dv/64, N2v/128, Bv), 256, 0, stream>>>(
        x2, Dv, (long long)N2v*Dv, lf_w, Dv, 0, lf_b,
        lf + (size_t)N1v*Dv, Dv, (long long)Sv*Dv, N2v, Dv, Dv);

    // 3) xout = elu(x1 @ af2_w.T + b)   (flat M = B*N1)
    gemm_wmma_kernel<1,true,false,false><<<dim3(Dv/64, (Bv*N1v)/128, 1), 256, 0, stream>>>(
        x1, Dv, 0, af2_w, Dv, 0, af2_b, xout, Dv, 0, Bv*N1v, Dv, Dv);

    // 4) xout += elu(af3_w @ x2[b] + af3_b[perM])  (batched, B not transposed)
    gemm_wmma_kernel<1,false,true,true><<<dim3(Dv/64, N1v/128, Bv), 256, 0, stream>>>(
        af3_w, N2v, 0, x2, Dv, (long long)N2v*Dv, af3_b,
        xout, Dv, (long long)N1v*Dv, N1v, Dv, N2v);

    // 5) qkv = z @ attn_in_w.T + b  (two batched launches)
    gemm_wmma_kernel<0,true,false,false><<<dim3(3*Dv/64, N1v/128, Bv), 256, 0, stream>>>(
        x1, Dv, (long long)N1v*Dv, attn_in_w, Dv, 0, attn_in_b,
        qkvb, 3*Dv, (long long)Sv*3*Dv, N1v, 3*Dv, Dv);
    gemm_wmma_kernel<0,true,false,false><<<dim3(3*Dv/64, N2v/128, Bv), 256, 0, stream>>>(
        x2, Dv, (long long)N2v*Dv, attn_in_w, Dv, 0, attn_in_b,
        qkvb + (size_t)N1v*3*Dv, 3*Dv, (long long)Sv*3*Dv, N2v, 3*Dv, Dv);

    // 6) attention
    flash_kernel<<<dim3(Sv/128, Bv*Hv), 256, 0, stream>>>(qkvb, obuf);

    // 7) o_proj = o @ attn_out_w.T + b  (flat M = B*S)
    gemm_wmma_kernel<0,true,false,false><<<dim3(Dv/64, (Bv*Sv)/128, 1), 256, 0, stream>>>(
        obuf, Dv, 0, attn_out_w, Dv, 0, attn_out_b, ofin, Dv, 0, Bv*Sv, Dv, Dv);

    // 8) weighted combine
    combine_gene_kernel<<<(Bv*(size_t)N1v*Dv)/256, 256, 0, stream>>>(lf, xout, ofin, x1, wrt, gene);
    combine_img_kernel <<<(Bv*(size_t)N2v*Dv)/256, 256, 0, stream>>>(lf, xout, ofin, x2, wrt, img);

    (void)in_sizes; (void)n_in; (void)out_size; (void)ws_size;
}
